// Attention_20220706030385
// MI455X (gfx1250) — compile-verified, matrix operand order checked by reading, not run on hardware
//
#include <hip/hip_runtime.h>
#include <cstdint>
#include <cstddef>

#define B_   2
#define S_   2048
#define D_   2048
#define NH   16
#define NKV  4
#define HD   128
#define NREP (NH / NKV)

typedef __attribute__((ext_vector_type(8)))  unsigned int v8u;
typedef __attribute__((ext_vector_type(8)))  float        v8f;
typedef __attribute__((ext_vector_type(16))) __bf16       v16bf;

// ---------------- scalar bf16 helpers ----------------
__device__ __forceinline__ unsigned short f2bf(float f) {
  unsigned int u = __builtin_bit_cast(unsigned int, f);
  u += 0x7fffu + ((u >> 16) & 1u);            // round-to-nearest-even
  return (unsigned short)(u >> 16);
}
__device__ __forceinline__ float bf2f(unsigned short h) {
  return __builtin_bit_cast(float, (unsigned int)h << 16);
}

// ---------------- gfx1250 async copy: global -> LDS (ASYNCcnt) ----------------
__device__ __forceinline__ void async_ld_b128(unsigned int lds_byte_addr,
                                              const void* gaddr) {
  asm volatile("global_load_async_to_lds_b128 %0, %1, off"
               :: "v"(lds_byte_addr), "v"((unsigned long long)gaddr)
               : "memory");
}
__device__ __forceinline__ void wait_async0() {
  asm volatile("s_wait_asynccnt 0x0" ::: "memory");
}

// Load a 16x32 bf16 fragment (A or B operand) from LDS in the CDNA5 WMMA
// striping: lanes 0-15 hold K={0..7,16..23} pairs, lanes 16-31 hold
// K={8..15,24..31}; row = lane&15. strideHalves must be a multiple of 8 so
// both reads are 16B-aligned ds_load_b128.
__device__ __forceinline__ v16bf ld_frag16x32(const unsigned short* p0,
                                              int strideHalves, int lane) {
  const int r  = lane & 15;
  const int kh = (lane >> 4) << 3;
  const unsigned short* p = p0 + r * strideHalves + kh;
  uint4 lo = *(const uint4*)(p);
  uint4 hi = *(const uint4*)(p + 16);
  v8u f;
  f[0] = lo.x; f[1] = lo.y; f[2] = lo.z; f[3] = lo.w;
  f[4] = hi.x; f[5] = hi.y; f[6] = hi.z; f[7] = hi.w;
  return __builtin_bit_cast(v16bf, f);
}

__device__ __forceinline__ v8f wmma_bf16(v16bf a, v16bf b, v8f c) {
  return __builtin_amdgcn_wmma_f32_16x16x32_bf16(false, a, false, b,
                                                 (short)0, c, false, false);
}

// =====================  fp32 -> bf16 bulk convert  =====================
__global__ void cvt_f32_bf16(const float* __restrict__ in,
                             unsigned short* __restrict__ out, int n4) {
  int i = blockIdx.x * blockDim.x + threadIdx.x;
  if (i >= n4) return;
  float4 v = ((const float4*)in)[i];
  unsigned int lo = (unsigned int)f2bf(v.x) | ((unsigned int)f2bf(v.y) << 16);
  unsigned int hi = (unsigned int)f2bf(v.z) | ((unsigned int)f2bf(v.w) << 16);
  uint2 r; r.x = lo; r.y = hi;
  ((uint2*)out)[i] = r;
}

// ============  GEMM: O[M,N] = X[M,K] * W[N,K]^T   (bf16 in, bf16/f32 out) ============
#define LSTR 72  // 64 + 8 halves pad: 144B row stride -> conflict-free b128 reads

template <bool BF16OUT>
__global__ __launch_bounds__(256) void gemm_bf16(
    const unsigned short* __restrict__ X, const unsigned short* __restrict__ W,
    void* __restrict__ Oo, int M, int N, int Kdim) {
  __shared__ unsigned short Xs[128 * LSTR];
  __shared__ unsigned short Ws[128 * LSTR];

  const int tid  = threadIdx.x;
  const int lane = tid & 31;
  const int warp = tid >> 5;
  const int wm   = warp & 3;   // 4 wave-rows of 32
  const int wn   = warp >> 2;  // 2 wave-cols of 64
  const int tM   = blockIdx.y * 128;
  const int tN   = blockIdx.x * 128;
  const unsigned int xsOff = (unsigned int)(size_t)&Xs[0];
  const unsigned int wsOff = (unsigned int)(size_t)&Ws[0];

  v8f acc[2][4];
#pragma unroll
  for (int mi = 0; mi < 2; ++mi)
#pragma unroll
    for (int ni = 0; ni < 4; ++ni)
#pragma unroll
      for (int j = 0; j < 8; ++j) acc[mi][ni][j] = 0.0f;

  for (int kb = 0; kb < Kdim; kb += 64) {
    if (kb) __syncthreads();  // previous tile fully consumed before overwrite
    // ---- stage 128x64 bf16 tiles of X and W via async DMA to LDS ----
#pragma unroll
    for (int it = 0; it < 4; ++it) {
      int i  = tid + it * 256;   // 1024 16B chunks: 128 rows x 8
      int r  = i >> 3;
      int c8 = (i & 7) << 3;
      async_ld_b128(xsOff + (unsigned)(r * LSTR + c8) * 2u,
                    X + (size_t)(tM + r) * Kdim + kb + c8);
      async_ld_b128(wsOff + (unsigned)(r * LSTR + c8) * 2u,
                    W + (size_t)(tN + r) * Kdim + kb + c8);
    }
    if (kb + 64 < Kdim) {  // hint next K-slab toward L2 (global_prefetch_b8)
      __builtin_prefetch(X + (size_t)(tM + (tid >> 1)) * Kdim + kb + 64, 0, 1);
      __builtin_prefetch(W + (size_t)(tN + (tid >> 1)) * Kdim + kb + 64, 0, 1);
    }
    wait_async0();
    __syncthreads();

#pragma unroll
    for (int ks = 0; ks < 2; ++ks) {
      v16bf afrag[2], bfrag[4];
#pragma unroll
      for (int mi = 0; mi < 2; ++mi)
        afrag[mi] = ld_frag16x32(&Xs[(wm * 32 + mi * 16) * LSTR + ks * 32], LSTR, lane);
#pragma unroll
      for (int ni = 0; ni < 4; ++ni)
        bfrag[ni] = ld_frag16x32(&Ws[(wn * 64 + ni * 16) * LSTR + ks * 32], LSTR, lane);
#pragma unroll
      for (int mi = 0; mi < 2; ++mi)
#pragma unroll
        for (int ni = 0; ni < 4; ++ni)
          acc[mi][ni] = wmma_bf16(afrag[mi], bfrag[ni], acc[mi][ni]);
    }
  }

  const int cn = lane & 15;
  const int rh = (lane >> 4) << 3;  // C layout: VGPR j -> row j (+8 for hi half)
#pragma unroll
  for (int mi = 0; mi < 2; ++mi)
#pragma unroll
    for (int ni = 0; ni < 4; ++ni)
#pragma unroll
      for (int j = 0; j < 8; ++j) {
        size_t idx = (size_t)(tM + wm * 32 + mi * 16 + j + rh) * N
                   + (tN + wn * 64 + ni * 16 + cn);
        if (BF16OUT) ((unsigned short*)Oo)[idx] = f2bf(acc[mi][ni][j]);
        else         ((float*)Oo)[idx]          = acc[mi][ni][j];
      }
}

// ==============  RoPE in-place on bf16 (optionally folds 1/sqrt(d))  ==============
__global__ void rope_bf16(unsigned short* __restrict__ T,
                          const float* __restrict__ C, const float* __restrict__ Sn,
                          int nheads, int rows, float outscale) {
  int idx   = blockIdx.x * blockDim.x + threadIdx.x;
  int total = rows * nheads * (HD / 2);
  if (idx >= total) return;
  int i  = idx & 63;
  int hh = (idx >> 6) % nheads;
  int rs = idx / (64 * nheads);
  int s  = rs & (S_ - 1);
  float c  = C[s * 64 + i];
  float sn = Sn[s * 64 + i];
  unsigned short* p = T + (size_t)rs * (size_t)(nheads * HD) + hh * HD + 2 * i;
  float e = bf2f(p[0]), o = bf2f(p[1]);
  p[0] = f2bf((e * c - o * sn) * outscale);
  p[1] = f2bf((e * sn + o * c) * outscale);
}

// ====================  Flash attention (GQA, causal, bf16 IO)  ====================
#define KSTR 136  // 128 + 8 halves pad
#define VSTR 40   // 32 + 8 halves pad
#define PSTR 40   // 32 + 8 halves pad (16B-aligned rows)

__global__ __launch_bounds__(128) void flash_attn(
    const unsigned short* __restrict__ Q, const unsigned short* __restrict__ Kc,
    const unsigned short* __restrict__ Vc, unsigned short* __restrict__ O) {
  __shared__ unsigned short Ks[32 * KSTR];   // [key][dim]
  __shared__ unsigned short Vt[HD * VSTR];   // [dim][key] (key pairs contiguous)
  __shared__ unsigned short Ps[4][16 * PSTR];

  const int tid  = threadIdx.x;
  const int lane = tid & 31;
  const int wid  = tid >> 5;
  const int qblk = blockIdx.x;
  const int h    = blockIdx.y;
  const int b    = blockIdx.z;
  const int g    = h / NREP;
  const int qbase = qblk * 64 + wid * 16;
  const unsigned int ksOff = (unsigned int)(size_t)&Ks[0];

  // ---- Q fragments straight from global bf16 (1/sqrt(d) pre-folded by RoPE) ----
  v16bf qf[4];
  {
    const int r  = lane & 15;
    const int kh = (lane >> 4) << 3;
    const unsigned short* qrow =
        Q + ((size_t)(b * S_) + qbase + r) * (size_t)D_ + h * HD;
#pragma unroll
    for (int c = 0; c < 4; ++c) {
      uint4 lo = *(const uint4*)(qrow + c * 32 + kh);
      uint4 hi = *(const uint4*)(qrow + c * 32 + 16 + kh);
      v8u f;
      f[0] = lo.x; f[1] = lo.y; f[2] = lo.z; f[3] = lo.w;
      f[4] = hi.x; f[5] = hi.y; f[6] = hi.z; f[7] = hi.w;
      qf[c] = __builtin_bit_cast(v16bf, f);
    }
  }

  v8f acc[8];
  float m[8], l[8];
#pragma unroll
  for (int c = 0; c < 8; ++c)
#pragma unroll
    for (int j = 0; j < 8; ++j) acc[c][j] = 0.0f;
#pragma unroll
  for (int j = 0; j < 8; ++j) { m[j] = -3.0e38f; l[j] = 0.0f; }

  const int kmax = qblk * 64 + 63;  // uniform trip count across the block
  for (int kb = 0; kb <= kmax; kb += 32) {
    __syncthreads();
    // ---- K tile: pure bf16 copy -> async DMA to LDS; V tile: transpose-stage ----
#pragma unroll
    for (int it = 0; it < 4; ++it) {
      int i   = tid + it * 128;  // 512 16B chunks: 32 keys x 16
      int key = i >> 4;
      int c8  = (i & 15) << 3;
      const size_t rowoff = ((size_t)(b * S_) + kb + key) * (size_t)(NKV * HD) + g * HD + c8;
      async_ld_b128(ksOff + (unsigned)(key * KSTR + c8) * 2u, Kc + rowoff);
      uint4 v = *(const uint4*)(Vc + rowoff);
      Vt[(c8 + 0) * VSTR + key] = (unsigned short)(v.x);
      Vt[(c8 + 1) * VSTR + key] = (unsigned short)(v.x >> 16);
      Vt[(c8 + 2) * VSTR + key] = (unsigned short)(v.y);
      Vt[(c8 + 3) * VSTR + key] = (unsigned short)(v.y >> 16);
      Vt[(c8 + 4) * VSTR + key] = (unsigned short)(v.z);
      Vt[(c8 + 5) * VSTR + key] = (unsigned short)(v.z >> 16);
      Vt[(c8 + 6) * VSTR + key] = (unsigned short)(v.w);
      Vt[(c8 + 7) * VSTR + key] = (unsigned short)(v.w >> 16);
    }
    wait_async0();
    __syncthreads();

    // ---- scores: two 16x16 tiles (keys kb..+15, kb+16..+31) ----
    v8f s0, s1;
#pragma unroll
    for (int j = 0; j < 8; ++j) { s0[j] = 0.0f; s1[j] = 0.0f; }
    {
      const int keyn = lane & 15;
#pragma unroll
      for (int c = 0; c < 4; ++c) {
        v16bf b0 = ld_frag16x32(&Ks[0 * KSTR + c * 32], KSTR, (lane & 15) | (lane & 16));
        // note: ld_frag16x32 uses lane&15 as row; keys 0..15 tile:
        b0 = ld_frag16x32(&Ks[c * 32], KSTR, lane);
        v16bf b1 = ld_frag16x32(&Ks[16 * KSTR + c * 32], KSTR, lane);
        s0 = wmma_bf16(qf[c], b0, s0);
        s1 = wmma_bf16(qf[c], b1, s1);
        (void)keyn;
      }
    }

    // ---- online softmax: row j lives in VGPR j across each 16-lane half ----
    const int keyn  = lane & 15;
    const int half8 = (lane >> 4) << 3;
#pragma unroll
    for (int j = 0; j < 8; ++j) {
      const int qr = qbase + j + half8;
      float a0 = s0[j] + ((kb + keyn)      > qr ? -1.0e9f : 0.0f);
      float a1 = s1[j] + ((kb + 16 + keyn) > qr ? -1.0e9f : 0.0f);
      float mx = fmaxf(a0, a1);
#pragma unroll
      for (int off = 1; off < 16; off <<= 1)
        mx = fmaxf(mx, __shfl_xor(mx, off, 16));
      const float nm    = fmaxf(m[j], mx);
      const float alpha = __expf(m[j] - nm);
      const float p0    = __expf(a0 - nm);
      const float p1    = __expf(a1 - nm);
      float rs = p0 + p1;
#pragma unroll
      for (int off = 1; off < 16; off <<= 1)
        rs += __shfl_xor(rs, off, 16);
      l[j] = l[j] * alpha + rs;
      m[j] = nm;
#pragma unroll
      for (int c = 0; c < 8; ++c) acc[c][j] *= alpha;
      unsigned short* pp = &Ps[wid][(j + half8) * PSTR];
      pp[keyn]      = f2bf(p0);
      pp[keyn + 16] = f2bf(p1);
    }
    __syncthreads();  // publish the P bounce for the A-layout reload

    // ---- P @ V : A = P (16q x 32 keys), B = V (32 keys x 16 dims) ----
    v16bf pf = ld_frag16x32(&Ps[wid][0], PSTR, lane);
    {
      const int dn = lane & 15;
#pragma unroll
      for (int c = 0; c < 8; ++c) {
        v16bf bv = ld_frag16x32(&Vt[(c * 16) * VSTR], VSTR, (lane & 16) | dn);
        acc[c] = wmma_bf16(pf, bv, acc[c]);
      }
    }
  }

  // ---- epilogue: normalize, write bf16 attention output ----
  const int dn    = lane & 15;
  const int half8 = (lane >> 4) << 3;
#pragma unroll
  for (int j = 0; j < 8; ++j) {
    const float inv  = 1.0f / l[j];
    const size_t row = (size_t)(b * S_) + qbase + j + half8;
#pragma unroll
    for (int c = 0; c < 8; ++c)
      O[row * (size_t)D_ + h * HD + c * 16 + dn] = f2bf(acc[c][j] * inv);
  }
}

// ================================  launcher  =================================
extern "C" void kernel_launch(void* const* d_in, const int* in_sizes, int n_in,
                              void* d_out, int out_size, void* d_ws, size_t ws_size,
                              hipStream_t stream) {
  (void)in_sizes; (void)n_in; (void)out_size; (void)ws_size;
  const float* x  = (const float*)d_in[0];
  const float* wq = (const float*)d_in[1];
  const float* wk = (const float*)d_in[2];
  const float* wv = (const float*)d_in[3];
  const float* wo = (const float*)d_in[4];
  const float* fc = (const float*)d_in[5];
  const float* fs = (const float*)d_in[6];
  // d_in[7] (mask) is reproduced analytically in the flash kernel
  float* out = (float*)d_out;

  const int M = B_ * S_;                       // 4096 token rows
  const size_t KVC = (size_t)NKV * HD;         // 512
  unsigned short* xb  = (unsigned short*)d_ws;          // M x 2048
  unsigned short* wqb = xb  + (size_t)M * D_;           // 2048 x 2048
  unsigned short* wkb = wqb + (size_t)D_ * D_;          // 512 x 2048
  unsigned short* wvb = wkb + KVC * D_;                 // 512 x 2048
  unsigned short* wob = wvb + KVC * D_;                 // 2048 x 2048
  unsigned short* Qb  = wob + (size_t)D_ * D_;          // M x 2048
  unsigned short* Kb  = Qb  + (size_t)M * D_;           // M x 512
  unsigned short* Vb  = Kb  + (size_t)M * KVC;          // M x 512
  unsigned short* Ab  = Vb  + (size_t)M * KVC;          // M x 2048

  const float qscale = 0.088388347648318447f;  // 1/sqrt(128)

  // one-time fp32 -> bf16 down-conversion (halves all downstream traffic)
  cvt_f32_bf16<<<(M * D_ / 4 + 255) / 256, 256, 0, stream>>>(x,  xb,  M * D_ / 4);
  cvt_f32_bf16<<<(D_ * D_ / 4 + 255) / 256, 256, 0, stream>>>(wq, wqb, D_ * D_ / 4);
  cvt_f32_bf16<<<((int)(KVC * D_) / 4 + 255) / 256, 256, 0, stream>>>(wk, wkb, (int)(KVC * D_) / 4);
  cvt_f32_bf16<<<((int)(KVC * D_) / 4 + 255) / 256, 256, 0, stream>>>(wv, wvb, (int)(KVC * D_) / 4);
  cvt_f32_bf16<<<(D_ * D_ / 4 + 255) / 256, 256, 0, stream>>>(wo, wob, D_ * D_ / 4);

  dim3 blk(256);
  gemm_bf16<true><<<dim3(D_ / 128, M / 128), blk, 0, stream>>>(xb, wqb, Qb, M, D_, D_);
  gemm_bf16<true><<<dim3((int)KVC / 128, M / 128), blk, 0, stream>>>(xb, wkb, Kb, M, (int)KVC, D_);
  gemm_bf16<true><<<dim3((int)KVC / 128, M / 128), blk, 0, stream>>>(xb, wvb, Vb, M, (int)KVC, D_);

  rope_bf16<<<(M * NH * 64 + 255) / 256, 256, 0, stream>>>(Qb, fc, fs, NH, M, qscale);
  rope_bf16<<<(M * NKV * 64 + 255) / 256, 256, 0, stream>>>(Kb, fc, fs, NKV, M, 1.0f);

  flash_attn<<<dim3(S_ / 64, NH, B_), dim3(128), 0, stream>>>(Qb, Kb, Vb, Ab);

  gemm_bf16<false><<<dim3(D_ / 128, M / 128), blk, 0, stream>>>(Ab, wob, out, M, D_, D_);
}
